// GraphTransformerRegressor_4243427688734
// MI455X (gfx1250) — matrix operand order repaired and checked by reading, not compiled, not run on hardware
//
#include <hip/hip_runtime.h>
#include <hip/hip_bf16.h>

// GraphTransformerRegressor for MI455X (gfx1250, wave32).
//
// Per layer:
//   1. qkvs_gemm : H[N,din] x {Wq,Wk,Wv,Ws}[din,64] (+bias) via v_wmma_f32_16x16x32_f16.
//      128-row panel per block staged as f16 in LDS (coalesced, padded stride,
//      16B-aligned ds_load_b128 fragments); B reused across 8 WMMAs per k-step.
//   2. init_layer: amax=-inf, denom=0, agg=0
//   3. edge_alpha: per (edge,head): e=edge_attr*We+be (recomputed, LDS-cached We),
//                  alpha=dot(q[dst], k[src]+e)/4, atomic-max into amax[dst,h]
//   4. edge_agg  : ex=exp(alpha-amax[dst]); atomicAdd denom; atomicAdd ex*(v[src]+e)
//   5. node_final: out = agg/denom + skip ; layernorm ; relu -> H
// Then mean-pool over graphs (atomics) + readout.
//
// Roofline: node tensors (12.8MB each) stay L2-resident (192MB L2) so the
// ~1.2GB/layer of random b128 gathers never hit HBM; HBM sees only streamed
// edge data (~1GB total -> ~45us at 23.3TB/s). e is recomputed per pass to
// avoid a 410MB/layer spill. GEMM FLOPs are tiny -> f16-input WMMA w/ f32 acc.

#define NNODES 50000
#define NEDGES 1600000
#define NGRAPH 64
#define HID    64
#define NHEAD  4
#define DHEAD  16

typedef _Float16 v16h __attribute__((ext_vector_type(16)));
typedef _Float16 v8h  __attribute__((ext_vector_type(8)));
typedef float    v8f  __attribute__((ext_vector_type(8)));

// ---------------------------------------------------------------------------
// Fused Q/K/V/S node GEMM.
// Block = 512 threads (16 waves) covers 128 rows x 256 cols (Q|K|V|S).
// Wave w -> matrix (w>>2), 16-col tile ((w&3)*16). 8 row-tiles per wave,
// B loaded once per k-step and reused across the 8 WMMAs.
// ---------------------------------------------------------------------------
__global__ void __launch_bounds__(512)
qkvs_gemm_kernel(const float* __restrict__ X, int din,
                 const float* __restrict__ Wq, const float* __restrict__ bq,
                 const float* __restrict__ Wk, const float* __restrict__ bk,
                 const float* __restrict__ Wv, const float* __restrict__ bv,
                 const float* __restrict__ Ws, const float* __restrict__ bs,
                 float* __restrict__ Q, float* __restrict__ Kout,
                 float* __restrict__ V, float* __restrict__ S)
{
    // Max din = 128; padded row stride (din+8 halves) keeps 16B alignment and
    // rotates LDS banks so 16 lanes x b128 reads are conflict-free.
    __shared__ __align__(16) _Float16 sA[128 * (128 + 8)];

    const int tid  = threadIdx.x;
    const int wave = tid >> 5;              // 0..15
    const int lane = tid & 31;
    const int hi   = lane >> 4;
    const int lo   = lane & 15;
    const int R0   = blockIdx.x * 128;
    const int mat  = wave >> 2;             // 0:Q 1:K 2:V 3:S
    const int col  = (wave & 3) * 16 + lo;  // 0..63 within matrix
    const int sas  = din + 8;               // LDS row stride in halves

    // Stage 128 x din A-panel, coalesced global reads, f32->f16 into LDS.
    for (int idx = tid; idx < 128 * din; idx += 512) {
        const int r = idx / din;
        const int k = idx - r * din;
        int row = R0 + r;
        if (row >= NNODES) row = NNODES - 1;   // clamp (stores are guarded)
        sA[r * sas + k] = (_Float16)X[(long long)row * din + k];
    }
    __syncthreads();

    const float* W; const float* B; float* OUT;
    switch (mat) {
        case 0:  W = Wq; B = bq; OUT = Q;    break;
        case 1:  W = Wk; B = bk; OUT = Kout; break;
        case 2:  W = Wv; B = bv; OUT = V;    break;
        default: W = Ws; B = bs; OUT = S;    break;
    }

    v8f acc[8];
    const float bval = B[col];
#pragma unroll
    for (int mt = 0; mt < 8; ++mt)
#pragma unroll
        for (int r = 0; r < 8; ++r) acc[mt][r] = bval;

    for (int k0 = 0; k0 < din; k0 += 32) {
        // B (32x16, column-per-lane): elem j -> K=k0+16*hi+j, N=lane&15
        v16h b;
        const float* wcol = W + (long long)(k0 + 16 * hi) * HID + col;
#pragma unroll
        for (int j = 0; j < 16; ++j) b[j] = (_Float16)wcol[j * HID];

#pragma unroll
        for (int mt = 0; mt < 8; ++mt) {
            // A (16x32, M=lane&15): elem j -> K=k0+16*hi+j
            const _Float16* ap = sA + (mt * 16 + lo) * sas + k0 + 16 * hi;
            const v8h c0 = *(const v8h*)ap;          // ds_load_b128
            const v8h c1 = *(const v8h*)(ap + 8);    // ds_load_b128
            v16h a;
#pragma unroll
            for (int j = 0; j < 8; ++j) { a[j] = c0[j]; a[8 + j] = c1[j]; }

            acc[mt] = __builtin_amdgcn_wmma_f32_16x16x32_f16(
                          /*neg_a=*/false, a, /*neg_b=*/false, b,
                          /*c_mod=*/(short)0, acc[mt],
                          /*reuse_a=*/false, /*reuse_b=*/false);
        }
    }

    // D layout: VGPR r -> row M = r + 8*hi, col N = lane&15
#pragma unroll
    for (int mt = 0; mt < 8; ++mt)
#pragma unroll
        for (int r = 0; r < 8; ++r) {
            const int row = R0 + mt * 16 + 8 * hi + r;
            if (row < NNODES) OUT[(long long)row * HID + col] = acc[mt][r];
        }
}

// ---------------------------------------------------------------------------
// Per-layer scratch init
// ---------------------------------------------------------------------------
__global__ void init_layer_kernel(float* __restrict__ amax,
                                  float* __restrict__ denom,
                                  float* __restrict__ agg)
{
    const int i = blockIdx.x * blockDim.x + threadIdx.x;
    if (i < NNODES * NHEAD) { amax[i] = -__builtin_inff(); denom[i] = 0.0f; }
    if (i < NNODES * HID)   agg[i] = 0.0f;
}

__device__ __forceinline__ void atomicMaxF(float* addr, float val)
{
    // Sign-split trick: non-negative floats order like ints; negative floats
    // order inversely as unsigned ints. Init value is -inf (0xFF800000).
    if (val >= 0.0f) atomicMax((int*)addr, __float_as_int(val));
    else             atomicMin((unsigned int*)addr, (unsigned int)__float_as_int(val));
}

__device__ __forceinline__ void load16(const float* __restrict__ p, float* v)
{
    // 64B-aligned 16-float slice -> 4x global_load_b128
    const float4* p4 = (const float4*)p;
#pragma unroll
    for (int i = 0; i < 4; ++i) {
        const float4 t = p4[i];
        v[4 * i + 0] = t.x; v[4 * i + 1] = t.y;
        v[4 * i + 2] = t.z; v[4 * i + 3] = t.w;
    }
}

// ---------------------------------------------------------------------------
// Edge pass 1: alpha + segment max.  One thread per (edge, head).
// ---------------------------------------------------------------------------
__global__ void __launch_bounds__(256)
edge_alpha_kernel(const float* __restrict__ EA,
                  const long long* __restrict__ EIDX,
                  const float* __restrict__ Q, const float* __restrict__ K,
                  const float* __restrict__ We, const float* __restrict__ be,
                  float* __restrict__ alpha, float* __restrict__ amax)
{
    __shared__ float sWe[16 * HID];
    __shared__ float sbe[HID];
    for (int i = threadIdx.x; i < 16 * HID; i += blockDim.x) sWe[i] = We[i];
    if (threadIdx.x < HID) sbe[threadIdx.x] = be[threadIdx.x];
    __syncthreads();

    const long long t = (long long)blockIdx.x * blockDim.x + threadIdx.x;
    const long long e = t >> 2;
    const int       h = (int)(t & 3);
    if (e >= NEDGES) return;

    const long long src = EIDX[e];
    const long long dst = EIDX[(long long)NEDGES + e];

    float ea[16];
    load16(EA + e * 16, ea);

    // e-projection for this head (16x16 MAC from LDS weights)
    float eh[16];
#pragma unroll
    for (int c2 = 0; c2 < 16; ++c2) eh[c2] = sbe[h * 16 + c2];
#pragma unroll
    for (int i = 0; i < 16; ++i) {
        const float av = ea[i];
#pragma unroll
        for (int c2 = 0; c2 < 16; ++c2) eh[c2] += av * sWe[i * HID + h * 16 + c2];
    }

    float kv[16], qv[16];
    load16(K + src * HID + h * 16, kv);   // L2-resident gathers
    load16(Q + dst * HID + h * 16, qv);

    float acc = 0.0f;
#pragma unroll
    for (int c2 = 0; c2 < 16; ++c2) acc += qv[c2] * (kv[c2] + eh[c2]);
    acc *= 0.25f;                         // 1/sqrt(D), D=16

    alpha[e * 4 + h] = acc;
    atomicMaxF(&amax[dst * 4 + h], acc);
}

// ---------------------------------------------------------------------------
// Edge pass 2: exp / denom / weighted-V aggregation (unnormalized).
// ---------------------------------------------------------------------------
__global__ void __launch_bounds__(256)
edge_agg_kernel(const float* __restrict__ EA,
                const long long* __restrict__ EIDX,
                const float* __restrict__ V,
                const float* __restrict__ We, const float* __restrict__ be,
                const float* __restrict__ alpha, const float* __restrict__ amax,
                float* __restrict__ denom, float* __restrict__ agg)
{
    __shared__ float sWe[16 * HID];
    __shared__ float sbe[HID];
    for (int i = threadIdx.x; i < 16 * HID; i += blockDim.x) sWe[i] = We[i];
    if (threadIdx.x < HID) sbe[threadIdx.x] = be[threadIdx.x];
    __syncthreads();

    const long long t = (long long)blockIdx.x * blockDim.x + threadIdx.x;
    const long long e = t >> 2;
    const int       h = (int)(t & 3);
    if (e >= NEDGES) return;

    const long long src = EIDX[e];
    const long long dst = EIDX[(long long)NEDGES + e];

    float ea[16];
    load16(EA + e * 16, ea);

    float eh[16];
#pragma unroll
    for (int c2 = 0; c2 < 16; ++c2) eh[c2] = sbe[h * 16 + c2];
#pragma unroll
    for (int i = 0; i < 16; ++i) {
        const float av = ea[i];
#pragma unroll
        for (int c2 = 0; c2 < 16; ++c2) eh[c2] += av * sWe[i * HID + h * 16 + c2];
    }

    const float ex = __expf(alpha[e * 4 + h] - amax[dst * 4 + h]);
    atomicAdd(&denom[dst * 4 + h], ex);

    float vv[16];
    load16(V + src * HID + h * 16, vv);
    float* ap = agg + dst * HID + h * 16;
#pragma unroll
    for (int c2 = 0; c2 < 16; ++c2)
        atomicAdd(&ap[c2], ex * (vv[c2] + eh[c2]));   // global_atomic_add_f32
}

// ---------------------------------------------------------------------------
// Per-node: normalize by denom, add skip, layernorm, relu.
// ---------------------------------------------------------------------------
__global__ void node_finalize_kernel(const float* __restrict__ agg,
                                     const float* __restrict__ denom,
                                     const float* __restrict__ S,
                                     const float* __restrict__ lng,
                                     const float* __restrict__ lnb,
                                     float* __restrict__ H)
{
    const int n = blockIdx.x * blockDim.x + threadIdx.x;
    if (n >= NNODES) return;

    float dh[NHEAD];
#pragma unroll
    for (int h = 0; h < NHEAD; ++h) dh[h] = denom[n * NHEAD + h] + 1e-16f;

    float av[HID], sv[HID], v[HID];
#pragma unroll
    for (int q = 0; q < 4; ++q) {
        load16(agg + (long long)n * HID + q * 16, av + q * 16);
        load16(S   + (long long)n * HID + q * 16, sv + q * 16);
    }

    float mu = 0.0f;
#pragma unroll
    for (int c = 0; c < HID; ++c) {
        const float x = av[c] / dh[c >> 4] + sv[c];
        v[c] = x;
        mu += x;
    }
    mu *= (1.0f / HID);
    float var = 0.0f;
#pragma unroll
    for (int c = 0; c < HID; ++c) { const float d = v[c] - mu; var += d * d; }
    var *= (1.0f / HID);
    const float rs = rsqrtf(var + 1e-5f);

    float4* out4 = (float4*)(H + (long long)n * HID);
#pragma unroll
    for (int q = 0; q < 16; ++q) {
        float4 o;
        o.x = fmaxf((v[4*q+0] - mu) * rs * lng[4*q+0] + lnb[4*q+0], 0.0f);
        o.y = fmaxf((v[4*q+1] - mu) * rs * lng[4*q+1] + lnb[4*q+1], 0.0f);
        o.z = fmaxf((v[4*q+2] - mu) * rs * lng[4*q+2] + lnb[4*q+2], 0.0f);
        o.w = fmaxf((v[4*q+3] - mu) * rs * lng[4*q+3] + lnb[4*q+3], 0.0f);
        out4[q] = o;
    }
}

// ---------------------------------------------------------------------------
// Global mean pool + readout
// ---------------------------------------------------------------------------
__global__ void init_pool_kernel(float* __restrict__ sums, float* __restrict__ counts)
{
    const int i = blockIdx.x * blockDim.x + threadIdx.x;
    if (i < NGRAPH * HID) sums[i] = 0.0f;
    if (i < NGRAPH)       counts[i] = 0.0f;
}

__global__ void pool_kernel(const float* __restrict__ H,
                            const long long* __restrict__ batch,
                            float* __restrict__ sums, float* __restrict__ counts)
{
    const long long t = (long long)blockIdx.x * blockDim.x + threadIdx.x;
    if (t >= (long long)NNODES * HID) return;
    const int n = (int)(t >> 6);
    const int c = (int)(t & 63);
    const int g = (int)batch[n];
    atomicAdd(&sums[g * HID + c], H[t]);
    if (c == 0) atomicAdd(&counts[g], 1.0f);
}

__global__ void readout_kernel(const float* __restrict__ sums,
                               const float* __restrict__ counts,
                               const float* __restrict__ rw,
                               const float* __restrict__ rb,
                               float* __restrict__ out)
{
    const int g = threadIdx.x;
    if (g >= NGRAPH) return;
    const float inv = 1.0f / fmaxf(counts[g], 1.0f);
    float acc = rb[0];
#pragma unroll
    for (int c = 0; c < HID; ++c) acc += sums[g * HID + c] * inv * rw[c];
    out[g] = acc;
}

// ---------------------------------------------------------------------------
// Host launcher
// ---------------------------------------------------------------------------
extern "C" void kernel_launch(void* const* d_in, const int* in_sizes, int n_in,
                              void* d_out, int out_size, void* d_ws, size_t ws_size,
                              hipStream_t stream)
{
    // Input flattening (dict insertion order, params nested in order):
    //  0: x [N,128]          1: edge_attr [E,16]
    //  per layer l (base = 2 + 12*l):
    //   +0 Wq +1 bq +2 Wk +3 bk +4 Wv +5 bv +6 We +7 be +8 Ws +9 bs +10 ln_g +11 ln_b
    //  26: readout_w [64]    27: readout_b [1]
    //  28: edge_index [2,E] (int64)   29: batch [N] (int64)
    const float*     x     = (const float*)d_in[0];
    const float*     ea    = (const float*)d_in[1];
    const float*     rw    = (const float*)d_in[26];
    const float*     rb    = (const float*)d_in[27];
    const long long* eidx  = (const long long*)d_in[28];
    const long long* batch = (const long long*)d_in[29];

    // Workspace layout (floats). Total ~26.0M floats (~104 MB).
    float* ws    = (float*)d_ws;
    float* Q     = ws;                    // N*64
    float* K     = Q     + (size_t)NNODES * HID;
    float* V     = K     + (size_t)NNODES * HID;
    float* S     = V     + (size_t)NNODES * HID;
    float* H     = S     + (size_t)NNODES * HID;   // layer output / next input
    float* alpha = H     + (size_t)NNODES * HID;   // E*4
    float* amax  = alpha + (size_t)NEDGES * NHEAD; // N*4
    float* denom = amax  + (size_t)NNODES * NHEAD; // N*4
    float* agg   = denom + (size_t)NNODES * NHEAD; // N*64
    float* sums  = agg   + (size_t)NNODES * HID;   // G*64
    float* cnts  = sums  + (size_t)NGRAPH * HID;   // G

    const dim3 gemmGrid((NNODES + 127) / 128), gemmBlk(512);
    const dim3 initGrid((NNODES * HID + 255) / 256), blk256(256);
    const dim3 edgeGrid(((size_t)NEDGES * NHEAD + 255) / 256);
    const dim3 nodeGrid((NNODES + 255) / 256);

    const float* Xin = x;
    int din = 128;
    for (int l = 0; l < 2; ++l) {
        const int pb = 2 + 12 * l;
        const float* Wq  = (const float*)d_in[pb + 0];
        const float* bq  = (const float*)d_in[pb + 1];
        const float* Wk  = (const float*)d_in[pb + 2];
        const float* bk  = (const float*)d_in[pb + 3];
        const float* Wv  = (const float*)d_in[pb + 4];
        const float* bv  = (const float*)d_in[pb + 5];
        const float* We  = (const float*)d_in[pb + 6];
        const float* be  = (const float*)d_in[pb + 7];
        const float* Wsk = (const float*)d_in[pb + 8];
        const float* bsk = (const float*)d_in[pb + 9];
        const float* lng = (const float*)d_in[pb + 10];
        const float* lnb = (const float*)d_in[pb + 11];

        qkvs_gemm_kernel<<<gemmGrid, gemmBlk, 0, stream>>>(
            Xin, din, Wq, bq, Wk, bk, Wv, bv, Wsk, bsk, Q, K, V, S);

        init_layer_kernel<<<initGrid, blk256, 0, stream>>>(amax, denom, agg);

        edge_alpha_kernel<<<edgeGrid, blk256, 0, stream>>>(
            ea, eidx, Q, K, We, be, alpha, amax);

        edge_agg_kernel<<<edgeGrid, blk256, 0, stream>>>(
            ea, eidx, V, We, be, alpha, amax, denom, agg);

        node_finalize_kernel<<<nodeGrid, blk256, 0, stream>>>(
            agg, denom, S, lng, lnb, H);

        Xin = H;
        din = HID;
    }

    init_pool_kernel<<<dim3((NGRAPH * HID + 255) / 256), blk256, 0, stream>>>(sums, cnts);
    pool_kernel<<<initGrid, blk256, 0, stream>>>(H, batch, sums, cnts);
    readout_kernel<<<dim3(1), dim3(64), 0, stream>>>(sums, cnts, rw, rb, (float*)d_out);
}
